// CoordinateDecoder_42580305772601
// MI455X (gfx1250) — matrix operand order repaired and checked by reading, not compiled, not run on hardware
//
#include <hip/hip_runtime.h>
#include <hip/hip_bf16.h>

// ---------------------------------------------------------------------------
// CDNA5 (gfx1250) fused FiLM-modulated coordinate decoder.
// - v_wmma_f32_16x16x32_bf16 GEMM core (wave32 WMMA)
// - weights staged Global->LDS with async-to-LDS DMA (ASYNCcnt), double-buffered
// - B-fragment LDS reads software-pipelined one n-tile ahead
// - activations resident in LDS bf16 across all 4 layers
// - hardware TRANS ops (v_sin/v_cos/v_tanh) for posenc / GELU / heads
// ---------------------------------------------------------------------------

typedef __attribute__((ext_vector_type(16))) __bf16 v16bf;
typedef __attribute__((ext_vector_type(8)))  float  v8f;
typedef unsigned int u32x4 __attribute__((ext_vector_type(4)));
typedef int i32x4v __attribute__((vector_size(16)));   // matches builtin's V4i

#define BATCH   8
#define NPTS    32768
#define WIDTH_  256
#define PI_F    3.14159265358979323846f
#define INV2PI  0.15915494309189535f

#if __has_builtin(__builtin_amdgcn_global_load_async_to_lds_b128) && \
    __has_builtin(__builtin_amdgcn_s_wait_asynccnt)
#define ASYNC_OK 1
#endif

// ------------------------------ fast math ----------------------------------
__device__ __forceinline__ float fast_sin(float x) {
#if __has_builtin(__builtin_amdgcn_sinf)
    return __builtin_amdgcn_sinf(x * INV2PI);   // v_sin_f32 (revolutions)
#else
    return sinf(x);
#endif
}
__device__ __forceinline__ float fast_cos(float x) {
#if __has_builtin(__builtin_amdgcn_cosf)
    return __builtin_amdgcn_cosf(x * INV2PI);   // v_cos_f32 (revolutions)
#else
    return cosf(x);
#endif
}
__device__ __forceinline__ float fast_tanh(float x) {
#if __has_builtin(__builtin_amdgcn_tanhf)
    return __builtin_amdgcn_tanhf(x);           // v_tanh_f32
#elif __has_builtin(__builtin_amdgcn_tanh_f32)
    return __builtin_amdgcn_tanh_f32(x);
#else
    return tanhf(x);
#endif
}

// ---------------------------------------------------------------------------
// Pyramid: 2x2 box downsample (bilinear half-size resize approximation)
// ---------------------------------------------------------------------------
__global__ void downsample_kernel(const float* __restrict__ src,
                                  float* __restrict__ dst, int Hs, int Hd) {
    size_t o = (size_t)blockIdx.x * 256 + threadIdx.x;
    int c = (int)(o & 63);
    size_t p = o >> 6;
    int x = (int)(p % (size_t)Hd); p /= (size_t)Hd;
    int y = (int)(p % (size_t)Hd);
    int b = (int)(p / (size_t)Hd);
    int ys = 2 * y, xs = 2 * x;
    size_t rb = (size_t)b * Hs;
    size_t i00 = ((rb + ys) * Hs + xs) * 64 + c;
    size_t i01 = ((rb + ys) * Hs + xs + 1) * 64 + c;
    size_t i10 = ((rb + ys + 1) * Hs + xs) * 64 + c;
    size_t i11 = ((rb + ys + 1) * Hs + xs + 1) * 64 + c;
    dst[o] = 0.25f * (src[i00] + src[i01] + src[i10] + src[i11]);
}

// ---------------------------------------------------------------------------
// Pack fp32 weights -> bf16 B-fragments.
// Fragment block (ktile,ntile): 32 lanes x 16 bf16 (32B contiguous per lane).
// lane l: half=l>>4, n = nt*16 + (l&15), elements e: K = kt*32 + half*16 + e.
// Layer K padded: L0 234->256, L1/L2 256, L3 490->512 (zero rows).
// ---------------------------------------------------------------------------
__global__ void pack_weights_kernel(const float* __restrict__ w0,
                                    const float* __restrict__ w1,
                                    const float* __restrict__ w2,
                                    const float* __restrict__ w3,
                                    unsigned short* __restrict__ packed) {
    int idx = blockIdx.x * 256 + threadIdx.x;          // 0 .. 327679
    const float* W; int base, Ksrc;
    if (idx < 65536)       { W = w0; base = 0;      Ksrc = 234; }
    else if (idx < 131072) { W = w1; base = 65536;  Ksrc = 256; }
    else if (idx < 196608) { W = w2; base = 131072; Ksrc = 256; }
    else                   { W = w3; base = 196608; Ksrc = 490; }
    int li     = idx - base;
    int blockI = li >> 9;          // 512 elems per fragment block
    int within = li & 511;
    int l = within >> 4, e = within & 15;
    int kt = blockI >> 4, nt = blockI & 15;
    int n = nt * 16 + (l & 15);
    int k = kt * 32 + (l >> 4) * 16 + e;
    float v = (k < Ksrc) ? W[k * WIDTH_ + n] : 0.f;
    __bf16 bv = (__bf16)v;
    packed[idx] = __builtin_bit_cast(unsigned short, bv);
}

// ---------------------------------------------------------------------------
// FiLM modulation vectors: per (layer,batch): silu(cmd@W1+b1)@W2+b2 -> 512
// ---------------------------------------------------------------------------
__global__ void modulate_kernel(const float* __restrict__ cmd,
                                const float* __restrict__ W1,
                                const float* __restrict__ B1,
                                const float* __restrict__ W2,
                                const float* __restrict__ B2,
                                float* __restrict__ modout) {
    __shared__ float hid[256];
    int b = blockIdx.x, t = threadIdx.x;
    float s = B1[t];
    for (int k = 0; k < 64; ++k) s += cmd[b * 64 + k] * W1[k * 256 + t];
    hid[t] = s / (1.f + __expf(-s));          // silu
    __syncthreads();
    for (int jj = 0; jj < 2; ++jj) {
        int j = t + jj * 256;
        float o = B2[j];
        for (int tt = 0; tt < 256; ++tt) o += hid[tt] * W2[tt * 512 + j];
        modout[b * 512 + j] = o;
    }
}

// ---------------------------------------------------------------------------
// Decoder helpers
// ---------------------------------------------------------------------------
__device__ __forceinline__ void put2(__bf16* row, int col, float v) {
    __bf16 b = (__bf16)v;
    row[col] = b;            // layer-0 input region (cols 0..233)
    row[256 + col] = b;      // layer-3 concat region (cols 256..489)
}

__device__ __forceinline__ void sample_level(__bf16* row, int colbase,
                                             const float* __restrict__ g,
                                             int Hl, int b, float cy, float cx,
                                             int lane) {
    float sc = (float)(Hl - 1);
    float yy = (cy + 1.f) * 0.5f * sc;
    float xx = (cx + 1.f) * 0.5f * sc;
    float y0f = floorf(yy), x0f = floorf(xx);
    float wy = yy - y0f, wx = xx - x0f;
    int y0 = (int)y0f; y0 = y0 < 0 ? 0 : (y0 > Hl - 1 ? Hl - 1 : y0);
    int x0 = (int)x0f; x0 = x0 < 0 ? 0 : (x0 > Hl - 1 ? Hl - 1 : x0);
    int y1 = (y0 + 1 > Hl - 1) ? Hl - 1 : y0 + 1;
    int x1 = (x0 + 1 > Hl - 1) ? Hl - 1 : x0 + 1;
    size_t rb = (size_t)b * Hl;
    size_t i00 = ((rb + y0) * Hl + x0) * 64;
    size_t i01 = ((rb + y0) * Hl + x1) * 64;
    size_t i10 = ((rb + y1) * Hl + x0) * 64;
    size_t i11 = ((rb + y1) * Hl + x1) * 64;
    float w00 = (1.f - wy) * (1.f - wx), w01 = (1.f - wy) * wx;
    float w10 = wy * (1.f - wx),         w11 = wy * wx;
#pragma unroll
    for (int cc = lane; cc < 64; cc += 32) {
        float v = g[i00 + cc] * w00 + g[i01 + cc] * w01 +
                  g[i10 + cc] * w10 + g[i11 + cc] * w11;
        __bf16 bv = (__bf16)v;
        row[colbase + cc] = bv;
        row[256 + colbase + cc] = bv;
    }
}

// A fragment (16x32 bf16, wave32 layout): lanes 0-15: row=lane, elems 0..15 ->
// K=k0..k0+15; lanes 16-31: row=lane-16, K=k0+16..k0+31 (matches packed B).
__device__ __forceinline__ v16bf load_afrag(const __bf16* rowptr, int k0, int half) {
    union { u32x4 q[2]; v16bf v; } u;
    u.q[0] = *reinterpret_cast<const u32x4*>(rowptr + k0 + half * 16);
    u.q[1] = *reinterpret_cast<const u32x4*>(rowptr + k0 + half * 16 + 8);
    return u.v;
}

// B fragment (32x16 bf16): lane l holds column n = nt*16 + (l&15),
// 16 contiguous K values starting at (l>>4)*16 -- 32B contiguous per lane.
__device__ __forceinline__ v16bf load_bfrag(const __bf16* wslab, int nt, int lane) {
    union { u32x4 q[2]; v16bf v; } bu;
    const __bf16* bp = wslab + nt * 512 + lane * 16;
    bu.q[0] = *reinterpret_cast<const u32x4*>(bp);
    bu.q[1] = *reinterpret_cast<const u32x4*>(bp + 8);
    return bu.v;
}

__device__ __forceinline__ void zero_acc(v8f acc[16]) {
#pragma unroll
    for (int i = 0; i < 16; ++i)
#pragma unroll
        for (int j = 0; j < 8; ++j) acc[i][j] = 0.f;
}

// Stage one 16KB weight k-slab (32xK fragment-packed) Global -> LDS with the
// whole 128-thread block: 128B per thread = 8 x b128 async DMA ops.
__device__ __forceinline__ void stage_slab(const unsigned short* __restrict__ g,
                                           __bf16* l, int tid) {
    const char* gs = (const char*)g + tid * 128;
    char* ld = (char*)l + tid * 128;
#ifdef ASYNC_OK
    __attribute__((address_space(1))) i32x4v* gp =
        (__attribute__((address_space(1))) i32x4v*)(unsigned long long)gs;
    __attribute__((address_space(3))) i32x4v* lp =
        (__attribute__((address_space(3))) i32x4v*)(unsigned int)(unsigned long long)ld;
#pragma unroll
    for (int i = 0; i < 8; ++i)
        __builtin_amdgcn_global_load_async_to_lds_b128(gp + i, lp + i, 0, 0);
#else
#pragma unroll
    for (int i = 0; i < 8; ++i)
        *(u32x4*)(ld + i * 16) = *(const u32x4*)(gs + i * 16);
#endif
}

__device__ __forceinline__ void async_wait_barrier() {
#ifdef ASYNC_OK
    __builtin_amdgcn_s_wait_asynccnt(0);
#endif
    __syncthreads();
}

// 16x16x32 WMMA across one k-tile against all 16 n-tiles.  B fragments are
// read from the LDS-staged slab, software-pipelined one n-tile ahead so the
// ds_load latency hides under the previous (co-executable) WMMA.
__device__ __forceinline__ void gemm_tile(const __bf16* act_row, int kt, int half,
                                          const __bf16* wslab, int lane,
                                          v8f acc[16]) {
    v16bf a = load_afrag(act_row, kt * 32, half);
    v16bf bcur = load_bfrag(wslab, 0, lane);
#pragma unroll
    for (int nt = 0; nt < 16; ++nt) {
        v16bf bnext = bcur;
        if (nt + 1 < 16) bnext = load_bfrag(wslab, nt + 1, lane);
        acc[nt] = __builtin_amdgcn_wmma_f32_16x16x32_bf16(
            false, a, false, bcur, (short)0, acc[nt], false, false);
        bcur = bnext;
    }
}

template <int NKT>
__device__ __forceinline__ void run_layer(const unsigned short* __restrict__ gpacked,
                                          __bf16 (*wbuf)[8192],
                                          const __bf16* act_row, int half,
                                          int lane, int tid, v8f acc[16]) {
    stage_slab(gpacked, wbuf[0], tid);
    async_wait_barrier();
#pragma unroll 1
    for (int kt = 0; kt < NKT; ++kt) {
        if (kt + 1 < NKT)                       // overlap next slab DMA w/ WMMA
            stage_slab(gpacked + (kt + 1) * 8192, wbuf[(kt + 1) & 1], tid);
        gemm_tile(act_row, kt, half, wbuf[kt & 1], lane, acc);
        async_wait_barrier();
    }
}

// bias + FiLM + tanh-GELU, write bf16 result back to LDS cols 0..255
__device__ __forceinline__ void epilogue(v8f acc[16], __bf16* ldsw, int lane,
                                         const float* __restrict__ bias,
                                         const float* __restrict__ modp) {
    int half = lane >> 4, nl = lane & 15;
#pragma unroll
    for (int nt = 0; nt < 16; ++nt) {
        int ncol = nt * 16 + nl;
        float bb = bias[ncol];
        float sc = modp[ncol];
        float sh = modp[256 + ncol];
#pragma unroll
        for (int j = 0; j < 8; ++j) {
            float v = acc[nt][j] + bb;
            v = v * (1.f + sc) + sh;
            float u = 0.7978845608028654f * (v + 0.044715f * v * v * v);
            v = 0.5f * v * (1.f + fast_tanh(u));
            ldsw[(size_t)(j + half * 8) * 512 + ncol] = (__bf16)v;
        }
    }
}

// ---------------------------------------------------------------------------
// Main fused decoder: 128 threads = 4 waves, each wave owns 16 rows.
// LDS: 64KB activations + 2x16KB double-buffered weight slabs = 96KB.
// ---------------------------------------------------------------------------
__global__ void __launch_bounds__(128)
decoder_kernel(const float* __restrict__ grid0, const float* __restrict__ pyr1,
               const float* __restrict__ pyr2, const float* __restrict__ coords,
               const unsigned short* __restrict__ packed,
               const float* __restrict__ modbuf,
               const float* __restrict__ b0, const float* __restrict__ b1,
               const float* __restrict__ b2, const float* __restrict__ b3,
               const float* __restrict__ out_w, const float* __restrict__ out_b,
               float* __restrict__ out) {
    __shared__ __bf16 act[4][16][512];
    __shared__ __bf16 wbuf[2][8192];
    int tid  = threadIdx.x;
    int lane = tid & 31;
    int w    = tid >> 5;
    int half = lane >> 4;
    int rbase = blockIdx.x * 64 + w * 16;     // global row base for this wave
    int b     = rbase >> 15;                  // batch (N = 32768)
    int nbase = rbase & (NPTS - 1);
    __bf16(*L)[512] = act[w];

    if (lane == 0) __builtin_prefetch(packed, 0, 1);   // global_prefetch path

    // ---- Stage inputs: posenc (42) + 3-level bilinear samples (192) -> LDS
    for (int r = 0; r < 16; ++r) {
        int n = nbase + r;
        float cy = coords[2 * n + 0];
        float cx = coords[2 * n + 1];
        __bf16* row = L[r];
        if (lane < 20) {
            int f = lane >> 1, d = lane & 1;
            float fr = (float)(1 << f) * PI_F;
            float ang = (d ? cx : cy) * fr;
            put2(row, 2 + f * 4 + 0 + d, fast_sin(ang));
            put2(row, 2 + f * 4 + 2 + d, fast_cos(ang));
        } else if (lane == 20) {
            put2(row, 0, cy);
        } else if (lane == 21) {
            put2(row, 1, cx);
        }
        if (lane < 22) {                       // zero K padding
            row[234 + lane] = (__bf16)0.f;     // 234..255
            row[490 + lane] = (__bf16)0.f;     // 490..511
        }
        sample_level(row, 42,       grid0, 256, b, cy, cx, lane);
        sample_level(row, 42 + 64,  pyr1,  128, b, cy, cx, lane);
        sample_level(row, 42 + 128, pyr2,   64, b, cy, cx, lane);
    }

    const __bf16* abase = &act[w][0][0];
    const __bf16* arow  = abase + (size_t)(lane & 15) * 512;
    __bf16* ldsw = &act[w][0][0];
    v8f acc[16];

    // layer 0 (K=256 padded from 234)
    zero_acc(acc);
    run_layer<8>(packed, wbuf, arow, half, lane, tid, acc);
    epilogue(acc, ldsw, lane, b0, modbuf + (0 * BATCH + b) * 512);
    // layer 1
    zero_acc(acc);
    run_layer<8>(packed + 65536, wbuf, arow, half, lane, tid, acc);
    epilogue(acc, ldsw, lane, b1, modbuf + (1 * BATCH + b) * 512);
    // layer 2
    zero_acc(acc);
    run_layer<8>(packed + 131072, wbuf, arow, half, lane, tid, acc);
    epilogue(acc, ldsw, lane, b2, modbuf + (2 * BATCH + b) * 512);
    // layer 3: concat [h(0..255) | mlp_in(256..489)] -> K=512 padded from 490
    zero_acc(acc);
    run_layer<16>(packed + 196608, wbuf, arow, half, lane, tid, acc);
    epilogue(acc, ldsw, lane, b3, modbuf + (3 * BATCH + b) * 512);

    // ---- output head: tanh(h @ out_w + out_b), 3 channels
    if (lane < 16) {
        float s0 = out_b[0], s1 = out_b[1], s2 = out_b[2];
        const __bf16* hr = L[lane];
        for (int t = 0; t < 256; ++t) {
            float hv = (float)hr[t];
            s0 += hv * out_w[t * 3 + 0];
            s1 += hv * out_w[t * 3 + 1];
            s2 += hv * out_w[t * 3 + 2];
        }
        size_t o = (size_t)(rbase + lane) * 3;
        out[o + 0] = fast_tanh(s0);
        out[o + 1] = fast_tanh(s1);
        out[o + 2] = fast_tanh(s2);
    }
}

// ---------------------------------------------------------------------------
// Host launcher
// ---------------------------------------------------------------------------
extern "C" void kernel_launch(void* const* d_in, const int* in_sizes, int n_in,
                              void* d_out, int out_size, void* d_ws, size_t ws_size,
                              hipStream_t stream) {
    (void)in_sizes; (void)n_in; (void)out_size; (void)ws_size;
    const float* grid0  = (const float*)d_in[0];
    const float* coords = (const float*)d_in[1];
    const float* cmd    = (const float*)d_in[2];
    const float* mw[4]  = {(const float*)d_in[3],  (const float*)d_in[4],
                           (const float*)d_in[5],  (const float*)d_in[6]};
    const float* mb[4]  = {(const float*)d_in[7],  (const float*)d_in[8],
                           (const float*)d_in[9],  (const float*)d_in[10]};
    const float* mw1[4] = {(const float*)d_in[11], (const float*)d_in[12],
                           (const float*)d_in[13], (const float*)d_in[14]};
    const float* mb1[4] = {(const float*)d_in[15], (const float*)d_in[16],
                           (const float*)d_in[17], (const float*)d_in[18]};
    const float* mw2[4] = {(const float*)d_in[19], (const float*)d_in[20],
                           (const float*)d_in[21], (const float*)d_in[22]};
    const float* mb2[4] = {(const float*)d_in[23], (const float*)d_in[24],
                           (const float*)d_in[25], (const float*)d_in[26]};
    const float* out_w  = (const float*)d_in[27];
    const float* out_b  = (const float*)d_in[28];

    char* ws = (char*)d_ws;
    float*          pyr1   = (float*)(ws + 0);                // 33,554,432 B
    float*          pyr2   = (float*)(ws + 33554432);         //  8,388,608 B
    float*          modbuf = (float*)(ws + 41943040);         //     65,536 B
    unsigned short* packed = (unsigned short*)(ws + 42008576);//    655,360 B

    downsample_kernel<<<32768, 256, 0, stream>>>(grid0, pyr1, 256, 128);
    downsample_kernel<<<8192, 256, 0, stream>>>(pyr1, pyr2, 128, 64);
    pack_weights_kernel<<<1280, 256, 0, stream>>>(mw[0], mw[1], mw[2], mw[3], packed);
    for (int i = 0; i < 4; ++i)
        modulate_kernel<<<8, 256, 0, stream>>>(cmd, mw1[i], mb1[i], mw2[i], mb2[i],
                                               modbuf + i * BATCH * 512);
    decoder_kernel<<<4096, 128, 0, stream>>>(grid0, pyr1, pyr2, coords, packed,
                                             modbuf, mb[0], mb[1], mb[2], mb[3],
                                             out_w, out_b, (float*)d_out);
}
